// SpaceNet_66855460929798
// MI455X (gfx1250) — compile-verified, matrix operand order checked by reading, not run on hardware
//
#include <hip/hip_runtime.h>

// ---------------------------------------------------------------------------
// Fully-fused NeRF MLP for gfx1250 (MI455X), wave32 + v_wmma_f32_16x16x32_f16.
// Per block: 64 rows (4 wave32 x 16 rows). Activations ping-pong in LDS (f16).
// Weights pre-converted to f16 (transposed, zero-padded) into d_ws, then
// double-buffer streamed into LDS with GLOBAL_LOAD_ASYNC_TO_LDS_B128
// (ASYNCcnt) overlapping the WMMA compute of the previous K-chunk.
// ---------------------------------------------------------------------------

typedef __attribute__((ext_vector_type(16))) _Float16 v16h;
typedef __attribute__((ext_vector_type(8)))  _Float16 v8h;
typedef __attribute__((ext_vector_type(8)))  float    v8f;

#define BS       128            // threads per block (4 wave32)
#define MTILE    64             // rows per block
#define ASTRIDE  328            // halves per activation row (conflict-free, 16B aligned)
#define WSTRIDE  40             // halves per staged weight row (32 + 8 pad, 16B aligned)
#define KC       32             // K-chunk size (one WMMA K-step)
#define NPTS     (2048*128)
#define RGB_ELEMS (NPTS*3)      // rgbs first in d_out, density after

// LDS partition (in halves)
#define ACTA_OFF 0
#define ACTB_OFF (MTILE*ASTRIDE)                 // 20992
#define WBUF_OFF (2*MTILE*ASTRIDE)               // 41984
#define WBUF_HALVES (256*WSTRIDE)                // 10240 per buffer
#define PES_OFF  (WBUF_OFF + 2*WBUF_HALVES)      // 62464
#define DES_OFF  (PES_OFF + MTILE*64)            // 66560
#define SMEM_HALVES (DES_OFF + 32)               // 66592
#define SMEM_BYTES  (SMEM_HALVES*2)              // 133184 bytes -> 2 blocks / WGP

// Pre-transposed f16 weight offsets in d_ws (halves): layout [N16][Kpad]
static const int WOFF[10] = {0, 16384, 81920, 147456, 212992, 294912,
                             360448, 425984, 430080, 471040};
#define WTOTAL 473088

// ---------------------------------------------------------------------------
// Prep: fp32 [K][N] row-major -> f16 transposed padded [N16][Kpad] in d_ws
// ---------------------------------------------------------------------------
__global__ void spacenet_prep_weights(
    const float* w0, const float* w1, const float* w2, const float* w3,
    const float* w4, const float* w5, const float* w6, const float* w7,
    const float* w8, const float* w9, _Float16* ws)
{
    const float* W[10] = {w0,w1,w2,w3,w4,w5,w6,w7,w8,w9};
    const int off[11] = {0,16384,81920,147456,212992,294912,360448,425984,
                         430080,471040,473088};
    const int Kp[10]  = {64,256,256,256,320,256,256,256,320,128};
    const int Nr[10]  = {256,256,256,256,256,256,256,1,128,3};
    const int Kr[10]  = {63,256,256,256,319,256,256,256,283,128};

    int idx = blockIdx.x * blockDim.x + threadIdx.x;
    if (idx >= WTOTAL) return;
    int L = 0;
    while (idx >= off[L+1]) L++;
    int r = idx - off[L];
    int n = r / Kp[L];
    int k = r % Kp[L];
    float v = (n < Nr[L] && k < Kr[L]) ? W[L][k * Nr[L] + n] : 0.0f;
    ws[idx] = (_Float16)v;
}

// ---------------------------------------------------------------------------
// Fragment load: two aligned 16-byte LDS reads -> v16h (8 VGPRs)
// A 16x32 f16 layout: lanes 0-15 row M=l, halves = K{0..7,16..23};
//                     lanes 16-31 row M=l-16, halves = K{8..15,24..31}.
// B mirrors this from the pre-transposed [N][K] weight tile.
// ---------------------------------------------------------------------------
__device__ __forceinline__ v16h load_frag(const _Float16* p) {
    v8h lo = *reinterpret_cast<const v8h*>(p);
    v8h hi = *reinterpret_cast<const v8h*>(p + 16);
    return __builtin_shufflevector(lo, hi, 0,1,2,3,4,5,6,7,8,9,10,11,12,13,14,15);
}

// ---------------------------------------------------------------------------
// Async-stage one 32-wide K-chunk of the [N16][Kpad] f16 weight panel into
// LDS via GLOBAL_LOAD_ASYNC_TO_LDS_B128 (no VGPR round-trip, ASYNCcnt).
// Each 16B segment: global = Wt + n*Kpad + c*KC + seg*8 halves.
// ---------------------------------------------------------------------------
__device__ __forceinline__ void stage_async(const _Float16* __restrict__ Wt,
                                            int Kpad, int c, _Float16* dst,
                                            int N16, int tid)
{
    const int NSEG = N16 * 4;            // 16-byte segments (32 halves / row)
    for (int i = tid; i < NSEG; i += BS) {
        int n   = i >> 2;
        int seg = (i & 3) * 8;           // halves within chunk row
        unsigned lds  = (unsigned)(size_t)(dst + n * WSTRIDE + seg);
        unsigned goff = (unsigned)((n * Kpad + c * KC + seg) * 2);   // bytes
        asm volatile("global_load_async_to_lds_b128 %0, %1, %2"
                     :: "v"(lds), "v"(goff), "s"(Wt) : "memory");
    }
}

__device__ __forceinline__ void wait_async_all() {
    asm volatile("s_wait_asynccnt 0x0" ::: "memory");
    __syncthreads();
}

// ---------------------------------------------------------------------------
// One GEMM layer: D(64 x NT*16) = act(64 x KCH*32) * W + bias, WMMA-tiled,
// double-buffered async weight streaming.
// MODE 0: relu -> actOut ; MODE 1: density -> out ; MODE 2: rgb -> out
// ---------------------------------------------------------------------------
template<int NT, int KCH, int MODE>
__device__ __forceinline__ void layer(
    const _Float16* __restrict__ actIn, _Float16* __restrict__ actOut,
    _Float16* __restrict__ wbuf, const _Float16* __restrict__ Wt,
    const float* __restrict__ Bias, int Nreal,
    float* __restrict__ out, int g0, int tid, int l16, int grp, int rowbase)
{
    const int Kpad = KCH * KC;
    const int N16  = NT * 16;
    v8f acc[NT];
#pragma unroll
    for (int nt = 0; nt < NT; nt++) {
        int n = nt * 16 + l16;
        float bv = (n < Nreal) ? Bias[n] : 0.0f;
#pragma unroll
        for (int j = 0; j < 8; j++) acc[nt][j] = bv;
    }

    const _Float16* arow = actIn + (rowbase + l16) * ASTRIDE + grp * 8;
    _Float16* wb[2] = {wbuf, wbuf + WBUF_HALVES};

    // prologue: stage chunk 0
    stage_async(Wt, Kpad, 0, wb[0], N16, tid);
    wait_async_all();

    for (int c = 0; c < KCH; c++) {
        if (c + 1 < KCH)                 // overlap: stream next chunk
            stage_async(Wt, Kpad, c + 1, wb[(c + 1) & 1], N16, tid);

        v16h a = load_frag(arow + c * KC);
        const _Float16* brow = wb[c & 1] + l16 * WSTRIDE + grp * 8;
#pragma unroll
        for (int nt = 0; nt < NT; nt++) {
            v16h b = load_frag(brow + nt * 16 * WSTRIDE);
            acc[nt] = __builtin_amdgcn_wmma_f32_16x16x32_f16(
                false, a, false, b, (short)0, acc[nt], false, false);
        }
        wait_async_all();                // next chunk landed; all waves synced
    }

    if (MODE == 0) {
        // D tile: VGPR r, lanes 0-15 -> M=r, lanes 16-31 -> M=r+8; N=l16
#pragma unroll
        for (int nt = 0; nt < NT; nt++) {
            int n = nt * 16 + l16;
#pragma unroll
            for (int r = 0; r < 8; r++) {
                int m = rowbase + grp * 8 + r;
                actOut[m * ASTRIDE + n] = (_Float16)fmaxf(acc[nt][r], 0.0f);
            }
        }
    } else if (MODE == 1) {              // density head (N=1, raw)
        if (l16 == 0) {
#pragma unroll
            for (int r = 0; r < 8; r++) {
                int g = g0 + rowbase + grp * 8 + r;
                out[RGB_ELEMS + g] = acc[0][r];
            }
        }
    } else {                             // rgb head (N=3, raw)
        if (l16 < 3) {
#pragma unroll
            for (int r = 0; r < 8; r++) {
                int g = g0 + rowbase + grp * 8 + r;
                out[3 * g + l16] = acc[0][r];
            }
        }
    }
}

// ---------------------------------------------------------------------------
// Fused forward kernel
// ---------------------------------------------------------------------------
__global__ void __launch_bounds__(BS, 1) spacenet_fused(
    const float* __restrict__ pos, const float* __restrict__ rays,
    const _Float16* wt0, const _Float16* wt1, const _Float16* wt2,
    const _Float16* wt3, const _Float16* wt4, const _Float16* wt5,
    const _Float16* wt6, const _Float16* wt7, const _Float16* wt8,
    const _Float16* wt9,
    const float* b0, const float* b1, const float* b2, const float* b3,
    const float* b4, const float* b5, const float* b6, const float* b7,
    const float* b8, const float* b9,
    float* __restrict__ out)
{
    extern __shared__ _Float16 smem[];
    _Float16* actA = smem + ACTA_OFF;
    _Float16* actB = smem + ACTB_OFF;
    _Float16* wbuf = smem + WBUF_OFF;
    _Float16* peS  = smem + PES_OFF;
    _Float16* deS  = smem + DES_OFF;

    const int tid = threadIdx.x;
    const int lane = tid & 31;
    const int wave = tid >> 5;
    const int l16 = lane & 15;
    const int grp = lane >> 4;
    const int rowbase = wave * 16;
    const int g0 = blockIdx.x * MTILE;

    // ---- positional encoding: one row per thread (rows 0..63) ----
    if (tid < MTILE) {
        int g = g0 + tid;
        float px = pos[3*g+0], py = pos[3*g+1], pz = pos[3*g+2];
        _Float16* ar = actA + tid * ASTRIDE;
        _Float16* pr = peS  + tid * 64;
        ar[0] = (_Float16)px; pr[0] = ar[0];
        ar[1] = (_Float16)py; pr[1] = ar[1];
        ar[2] = (_Float16)pz; pr[2] = ar[2];
        float f = 1.0f;
#pragma unroll
        for (int fl = 0; fl < 10; fl++) {
            int cbase = 3 + fl * 6;
            float vals[6];
            vals[0] = __sinf(f*px); vals[1] = __sinf(f*py); vals[2] = __sinf(f*pz);
            vals[3] = __cosf(f*px); vals[4] = __cosf(f*py); vals[5] = __cosf(f*pz);
#pragma unroll
            for (int q = 0; q < 6; q++) {
                _Float16 h = (_Float16)vals[q];
                ar[cbase+q] = h; pr[cbase+q] = h;
            }
            f *= 2.0f;
        }
        ar[63] = (_Float16)0.0f; pr[63] = (_Float16)0.0f;   // K pad
    } else if (tid < MTILE + 32) {
        // ---- direction encoding (shared by all 64 rows of this block) ----
        int j = tid - MTILE;
        int nray = blockIdx.x >> 1;      // 128 samples/ray, 64 rows/block
        float dx = rays[6*nray+0], dy = rays[6*nray+1], dz = rays[6*nray+2];
        float inv = rsqrtf(dx*dx + dy*dy + dz*dz);
        dx *= inv; dy *= inv; dz *= inv;
        float val = 0.0f;
        if (j < 3) {
            val = (j == 0) ? dx : ((j == 1) ? dy : dz);
        } else if (j < 27) {
            int q = j - 3, fl = q / 6, w = q % 6;
            float f = (float)(1 << fl);
            float c = (w % 3 == 0) ? dx : ((w % 3 == 1) ? dy : dz);
            val = (w < 3) ? __sinf(f*c) : __cosf(f*c);
        }
        deS[j] = (_Float16)fmaxf(val, 0.0f);     // relu(concat) folds onto de
    }
    __syncthreads();

    // ---- stage 1 MLP ----
    layer<16,2,0>(actA, actB, wbuf, wt0, b0, 256, out, g0, tid, l16, grp, rowbase);
    layer<16,8,0>(actB, actA, wbuf, wt1, b1, 256, out, g0, tid, l16, grp, rowbase);
    layer<16,8,0>(actA, actB, wbuf, wt2, b2, 256, out, g0, tid, l16, grp, rowbase);
    layer<16,8,0>(actB, actA, wbuf, wt3, b3, 256, out, g0, tid, l16, grp, rowbase);

    // concat [x, pe] into actA cols 256..319 (col 319 = 0 from peS pad)
    __syncthreads();
    for (int i = tid; i < MTILE * 64; i += BS) {
        int rw = i >> 6, c = i & 63;
        actA[rw * ASTRIDE + 256 + c] = peS[rw * 64 + c];
    }

    // ---- stage 2 MLP ----
    layer<16,10,0>(actA, actB, wbuf, wt4, b4, 256, out, g0, tid, l16, grp, rowbase);
    layer<16,8,0>(actB, actA, wbuf, wt5, b5, 256, out, g0, tid, l16, grp, rowbase);
    layer<16,8,0>(actA, actB, wbuf, wt6, b6, 256, out, g0, tid, l16, grp, rowbase); // stores relu(x)

    // append relu(de) to actB cols 256..282, zero 283..319
    __syncthreads();
    for (int i = tid; i < MTILE * 64; i += BS) {
        int rw = i >> 6, c = i & 63;
        actB[rw * ASTRIDE + 256 + c] = (c < 32) ? deS[c] : (_Float16)0.0f;
    }

    // ---- heads ----
    layer<1,8,1> (actB, nullptr, wbuf, wt7, b7, 1,   out, g0, tid, l16, grp, rowbase); // density
    layer<8,10,0>(actB, actA,    wbuf, wt8, b8, 128, out, g0, tid, l16, grp, rowbase); // rgb hidden
    layer<1,4,2> (actA, nullptr, wbuf, wt9, b9, 3,   out, g0, tid, l16, grp, rowbase); // rgbs
}

// ---------------------------------------------------------------------------
extern "C" void kernel_launch(void* const* d_in, const int* in_sizes, int n_in,
                              void* d_out, int out_size, void* d_ws, size_t ws_size,
                              hipStream_t stream) {
    (void)in_sizes; (void)n_in; (void)out_size; (void)ws_size;

    const float* pos  = (const float*)d_in[0];
    const float* rays = (const float*)d_in[1];
    const float* W[10];
    const float* B[10];
    for (int i = 0; i < 10; i++) {
        W[i] = (const float*)d_in[2 + 2*i];
        B[i] = (const float*)d_in[3 + 2*i];
    }
    _Float16* ws = (_Float16*)d_ws;

    // 1) weights -> f16, transposed + padded, into workspace (L2-resident)
    spacenet_prep_weights<<<(WTOTAL + 255) / 256, 256, 0, stream>>>(
        W[0],W[1],W[2],W[3],W[4],W[5],W[6],W[7],W[8],W[9], ws);

    // 2) fused MLP, 133 KB dynamic LDS per block (2 blocks / WGP)
    hipFuncSetAttribute((const void*)spacenet_fused,
                        hipFuncAttributeMaxDynamicSharedMemorySize, SMEM_BYTES);
    spacenet_fused<<<NPTS / MTILE, BS, SMEM_BYTES, stream>>>(
        pos, rays,
        ws + WOFF[0], ws + WOFF[1], ws + WOFF[2], ws + WOFF[3], ws + WOFF[4],
        ws + WOFF[5], ws + WOFF[6], ws + WOFF[7], ws + WOFF[8], ws + WOFF[9],
        B[0],B[1],B[2],B[3],B[4],B[5],B[6],B[7],B[8],B[9],
        (float*)d_out);
}
